// FusedExperts_29197187678926
// MI455X (gfx1250) — compile-verified, matrix operand order checked by reading, not run on hardware
//
#include <hip/hip_runtime.h>
#include <hip/hip_bf16.h>

// ---------------------------------------------------------------------------
// Fused MoE FFN (top-2 of 8 experts), bf16 WMMA with f32 accumulation.
//   out[n,:] += combine[n,e] * (relu(x[n,:] @ w1[e]^T)^2) @ w2[e]^T
// Strategy:
//   1) route (count / prefix / scatter) slot indices per expert
//   2) convert x, w1, w2 to bf16 in workspace (needs ~81 MB of d_ws)
//   3) grouped-GEMM blocks: 32 (token,slot) rows per block, one expert each,
//      v_wmma_f32_16x16x32_bf16 for both the up- and down-projection,
//      per-token weight folded into the hidden activations,
//      results atomically accumulated into zero-initialized d_out.
// ---------------------------------------------------------------------------

#define NTOK   8192
#define DDIM   1024
#define HDIM   2048
#define NEXP   8
#define TOPK   2
#define TILE_M 32
#define CHUNK_H 128
#define MAX_TILES ((NTOK * TOPK + TILE_M - 1) / TILE_M)   // 512

#define XSS (DDIM + 8)      // padded LDS stride (bf16 elems) -> conflict-free b128
#define HSS (CHUNK_H + 8)

typedef __attribute__((ext_vector_type(16))) __bf16 v16bf;
typedef __attribute__((ext_vector_type(8)))  __bf16 bf16x8;
typedef __attribute__((ext_vector_type(8)))  float  v8f;

union FragU { v16bf v; bf16x8 h[2]; };

// Load a 16x32 bf16 fragment in the CDNA5 A/B layout from an N-major (row =
// matrix row for A, matrix column for B), K-contiguous array with `stride`
// elements per row. Lane l (l<16) holds row l, K = 0..7 & 16..23; lanes 16..31
// hold row l-16, K = 8..15 & 24..31 (ISA 7.12.2).
__device__ __forceinline__ v16bf load_frag_nk(const __bf16* base, int stride, int lane) {
  const int row = lane & 15;
  const int sel = (lane >> 4) & 1;
  const __bf16* p = base + row * stride + sel * 8;
  FragU u;
  u.h[0] = *(const bf16x8*)(p);
  u.h[1] = *(const bf16x8*)(p + 16);
  return u.v;
}

__device__ __forceinline__ v8f wmma_bf16(v16bf a, v16bf b, v8f c) {
  return __builtin_amdgcn_wmma_f32_16x16x32_bf16(false, a, false, b, (short)0, c,
                                                 false, false);
}

// ------------------------------- small kernels ------------------------------

__global__ void k_zero_f32(float* p, int n) {
  int i = blockIdx.x * blockDim.x + threadIdx.x;
  int s = gridDim.x * blockDim.x;
  for (; i < n; i += s) p[i] = 0.0f;
}

__global__ void k_zero_int(int* p, int n) {
  int i = blockIdx.x * blockDim.x + threadIdx.x;
  if (i < n) p[i] = 0;
}

__global__ void k_f32_to_bf16(const float* __restrict__ src, __bf16* __restrict__ dst, int n) {
  int i = blockIdx.x * blockDim.x + threadIdx.x;
  int s = gridDim.x * blockDim.x;
  for (; i < n; i += s) dst[i] = (__bf16)src[i];
}

__global__ void k_count(const int* __restrict__ idx, int n, int* __restrict__ cnt) {
  int i = blockIdx.x * blockDim.x + threadIdx.x;
  int s = gridDim.x * blockDim.x;
  for (; i < n; i += s) atomicAdd(&cnt[idx[i] & (NEXP - 1)], 1);
}

__global__ void k_offsets(const int* __restrict__ cnt, int* __restrict__ off) {
  if (threadIdx.x == 0) {
    int run = 0;
    for (int e = 0; e < NEXP; ++e) { off[e] = run; run += cnt[e]; }
  }
}

__global__ void k_scatter(const int* __restrict__ idx, int n,
                          const int* __restrict__ off, int* __restrict__ fill,
                          int* __restrict__ perm) {
  int i = blockIdx.x * blockDim.x + threadIdx.x;
  int s = gridDim.x * blockDim.x;
  for (; i < n; i += s) {
    int e = idx[i] & (NEXP - 1);
    int pos = off[e] + atomicAdd(&fill[e], 1);
    perm[pos] = i;                       // slot id: token = i / TOPK
  }
}

// ------------------------------- main kernel --------------------------------

__global__ __launch_bounds__(256)
void k_moe(const __bf16* __restrict__ xb, const __bf16* __restrict__ w1b,
           const __bf16* __restrict__ w2b, const float* __restrict__ ew,
           const int* __restrict__ cnt, const int* __restrict__ off,
           const int* __restrict__ perm, float* __restrict__ out) {
  __shared__ __bf16 xs[TILE_M * XSS];   // gathered x tile (bf16)      ~66 KB
  __shared__ __bf16 hs[TILE_M * HSS];   // hidden chunk (bf16)          ~9 KB
  __shared__ int    tok[TILE_M];
  __shared__ float  wgt[TILE_M];

  const int e = blockIdx.x / MAX_TILES;
  const int t = blockIdx.x % MAX_TILES;
  const int n_e = cnt[e];
  if (t * TILE_M >= n_e) return;        // uniform exit: EXEC stays all-ones

  const int tid  = threadIdx.x;
  const int lane = tid & 31;
  const int wv   = tid >> 5;            // wave 0..7

  if (tid < TILE_M) {
    int r = t * TILE_M + tid;
    int p = (r < n_e) ? perm[off[e] + r] : perm[off[e] + t * TILE_M];
    tok[tid] = p / TOPK;
    wgt[tid] = (r < n_e) ? ew[p] : 0.0f;   // padded rows contribute 0
  }
  __syncthreads();

  // Gather 32 x 1024 bf16 x-tile (16-byte vectors, coalesced).
  for (int c = tid; c < TILE_M * (DDIM / 8); c += 256) {
    int row = c >> 7;                     // DDIM/8 == 128
    int c8  = c & 127;
    bf16x8 v = *(const bf16x8*)(xb + (size_t)tok[row] * DDIM + c8 * 8);
    *(bf16x8*)(&xs[row * XSS + c8 * 8]) = v;
  }
  __syncthreads();

  const __bf16* w1e = w1b + (size_t)e * HDIM * DDIM;   // (H, D) row-major
  const __bf16* w2e = w2b + (size_t)e * DDIM * HDIM;   // (D, H) row-major

  v8f acc[2][8];                        // wave owns 32 x 128 of the output
#pragma unroll
  for (int mh = 0; mh < 2; ++mh)
#pragma unroll
    for (int nt = 0; nt < 8; ++nt) acc[mh][nt] = (v8f){0.f,0.f,0.f,0.f,0.f,0.f,0.f,0.f};

  const int cl = lane & 15;
  const int rb = (lane >> 4) * 8;       // C-fragment row base (ISA 7.12.2)

  for (int hc = 0; hc < HDIM; hc += CHUNK_H) {
    // ---- phase A: hidden[32 x 128] chunk; wave wv computes h-cols wv*16..+15
    v8f ha0 = (v8f){0.f,0.f,0.f,0.f,0.f,0.f,0.f,0.f};
    v8f ha1 = ha0;
    const __bf16* w1t = w1e + (size_t)(hc + wv * 16) * DDIM;
#pragma unroll 4
    for (int d = 0; d < DDIM; d += 32) {
      v16bf bf = load_frag_nk(w1t + d, DDIM, lane);       // global b128 loads
      v16bf a0 = load_frag_nk(&xs[d], XSS, lane);          // ds b128 loads
      v16bf a1 = load_frag_nk(&xs[16 * XSS + d], XSS, lane);
      ha0 = wmma_bf16(a0, bf, ha0);
      ha1 = wmma_bf16(a1, bf, ha1);
    }
    // relu^2, fold per-token gate weight into hidden rows, store bf16 chunk
    const int hcb = wv * 16 + cl;
#pragma unroll
    for (int i = 0; i < 8; ++i) {
      int r0 = rb + i;
      float v0 = ha0[i]; v0 = fmaxf(v0, 0.f); v0 = v0 * v0 * wgt[r0];
      hs[r0 * HSS + hcb] = (__bf16)v0;
      int r1 = 16 + rb + i;
      float v1 = ha1[i]; v1 = fmaxf(v1, 0.f); v1 = v1 * v1 * wgt[r1];
      hs[r1 * HSS + hcb] = (__bf16)v1;
    }
    __syncthreads();

    // ---- phase B: acc += hidden_chunk @ w2^T, wave wv owns d-cols wv*128..+127
    const __bf16* w2t = w2e + (size_t)(wv * 128) * HDIM + hc;
#pragma unroll
    for (int kk = 0; kk < CHUNK_H; kk += 32) {
      v16bf a0 = load_frag_nk(&hs[kk], HSS, lane);
      v16bf a1 = load_frag_nk(&hs[16 * HSS + kk], HSS, lane);
#pragma unroll
      for (int nt = 0; nt < 8; ++nt) {
        v16bf bf = load_frag_nk(w2t + (size_t)(nt * 16) * HDIM + kk, HDIM, lane);
        acc[0][nt] = wmma_bf16(a0, bf, acc[0][nt]);
        acc[1][nt] = wmma_bf16(a1, bf, acc[1][nt]);
      }
    }
    __syncthreads();
  }

  // ---- accumulate the 32 x 1024 tile into out (token may appear in 2 slots)
#pragma unroll
  for (int mh = 0; mh < 2; ++mh)
#pragma unroll
    for (int nt = 0; nt < 8; ++nt) {
      int col = wv * 128 + nt * 16 + cl;
#pragma unroll
      for (int i = 0; i < 8; ++i) {
        int row = mh * 16 + rb + i;
        unsafeAtomicAdd(&out[(size_t)tok[row] * DDIM + col], acc[mh][nt][i]);
      }
    }
}

// ------------------------------- launcher -----------------------------------

extern "C" void kernel_launch(void* const* d_in, const int* in_sizes, int n_in,
                              void* d_out, int out_size, void* d_ws, size_t ws_size,
                              hipStream_t stream) {
  const float* x   = (const float*)d_in[0];
  const int*   idx = (const int*)d_in[1];   // expert_indices (per harness: int)
  const float* ew  = (const float*)d_in[2];
  const float* w1  = (const float*)d_in[3];
  const float* w2  = (const float*)d_in[4];
  float* out = (float*)d_out;

  // Workspace layout (requires ~81 MB):
  //   [0)      cnt[8]
  //   [32)     fill[8]
  //   [64)     off[8]
  //   [128)    perm[16384]
  //   [128K)   x_bf16   16 MB
  //            w1_bf16  32 MB
  //            w2_bf16  32 MB
  char* ws = (char*)d_ws;
  int* cnt  = (int*)(ws + 0);
  int* fill = (int*)(ws + 32);
  int* off  = (int*)(ws + 64);
  int* perm = (int*)(ws + 128);
  __bf16* xb  = (__bf16*)(ws + (1 << 17));
  __bf16* w1b = xb  + (size_t)NTOK * DDIM;
  __bf16* w2b = w1b + (size_t)NEXP * HDIM * DDIM;

  k_zero_f32<<<2048, 256, 0, stream>>>(out, NTOK * DDIM);
  k_zero_int<<<1, 32, 0, stream>>>(cnt, 16);                    // cnt + fill
  k_f32_to_bf16<<<4096, 256, 0, stream>>>(x,  xb,  NTOK * DDIM);
  k_f32_to_bf16<<<4096, 256, 0, stream>>>(w1, w1b, NEXP * HDIM * DDIM);
  k_f32_to_bf16<<<4096, 256, 0, stream>>>(w2, w2b, NEXP * DDIM * HDIM);
  k_count<<<64, 256, 0, stream>>>(idx, NTOK * TOPK, cnt);
  k_offsets<<<1, 1, 0, stream>>>(cnt, off);
  k_scatter<<<64, 256, 0, stream>>>(idx, NTOK * TOPK, off, fill, perm);
  k_moe<<<NEXP * MAX_TILES, 256, 0, stream>>>(xb, w1b, w2b, ew, cnt, off, perm, out);
}
